// SeqLabelLSTM_25426206392927
// MI455X (gfx1250) — matrix operand order repaired and checked by reading, not compiled, not run on hardware
//
#include <hip/hip_runtime.h>
#include <hip/hip_bf16.h>
#include <cstddef>
#include <cstdint>

typedef __attribute__((ext_vector_type(16))) _Float16 v16h;
typedef __attribute__((ext_vector_type(8)))  _Float16 v8h;
typedef __attribute__((ext_vector_type(4)))  _Float16 v4h;
typedef __attribute__((ext_vector_type(8)))  float    v8f;
typedef unsigned int u32x4 __attribute__((ext_vector_type(4)));
typedef int          i32x8 __attribute__((ext_vector_type(8)));
typedef int          i32x4 __attribute__((ext_vector_type(4)));

#define B_    256
#define T_    512
#define E_    64
#define H_    128
#define G4_   512     // 4*H
#define VOUT_ 128

// dynamic LDS layout (f16 elements)
#define LWH0_  0                  // Wh0: 512*128
#define LWH1_  65536              // Wh1: 512*128
#define LX_    131072             // x_t tile: 16*64
#define LH0_   132096             // h0: 16*128
#define LH1_   134144             // h1: 16*128
#define LDS_ELEMS_ 136192
#define LDS_BYTES_ (LDS_ELEMS_ * 2)

// ---------------------------------------------------------------------------
// WMMA helpers (CDNA5 wave32, V_WMMA_F32_16X16X32_F16)
// ---------------------------------------------------------------------------
static __device__ inline v8f wmma(v16h a, v16h b, v8f c) {
    return __builtin_amdgcn_wmma_f32_16x16x32_f16(false, a, false, b, (short)0, c, false, false);
}

static __device__ inline v16h combine16(v8h lo, v8h hi) {
    v16h r;
#pragma unroll
    for (int i = 0; i < 8; ++i) { r[i] = lo[i]; r[8 + i] = hi[i]; }
    return r;
}

// A-fragment (16x32 f16, MxK) from a row-major [16 x ld] f16 buffer (LDS or global).
static __device__ inline v16h loadA(const _Float16* base, int ld, int k0, int lane) {
    int m  = lane & 15;
    int kh = (lane >> 4) << 4;                 // 0 or 16
    const _Float16* p = base + m * ld + k0 + kh;
    v8h lo = *(const v8h*)(p);
    v8h hi = *(const v8h*)(p + 8);
    return combine16(lo, hi);
}

// B-fragment (32x16 f16, KxN), B = W^T with W row-major [N_total x ldK]:
// per lane one contiguous 32-byte read.
static __device__ inline v16h loadB_WT(const _Float16* W, int ldK, int nbase, int k0, int lane) {
    int n  = lane & 15;
    int kb = (lane >> 4) << 4;                 // 0 or 16
    return *(const v16h*)(W + (size_t)(nbase + n) * ldK + k0 + kb);
}

static __device__ inline v8f splat8(float x) {
    v8f v;
#pragma unroll
    for (int r = 0; r < 8; ++r) v[r] = x;
    return v;
}

static __device__ inline float sigmoidf_(float x) { return 1.0f / (1.0f + __expf(-x)); }

// ---------------------------------------------------------------------------
// TDM: bulk-copy a row-major 2D f16 matrix (rows x cols, contiguous) into LDS.
// Descriptor per CDNA5 ISA ch.8 (group0: count/lds_addr/global_addr/type,
// group1: data_size/dims/tile/stride). Tracked by TENSORcnt.
// This toolchain exposes the 6-arg builtin: (g0, g1, g2, g3, g2b, cpol).
// ---------------------------------------------------------------------------
static __device__ inline void tdm_load_2d_f16(unsigned lds_addr, const void* gptr,
                                              unsigned rows, unsigned cols) {
    unsigned long long ga = (unsigned long long)(uintptr_t)gptr;
    u32x4 g0;
    g0[0] = 1u;                                               // count=1, user desc
    g0[1] = lds_addr;                                         // lds_addr [63:32]
    g0[2] = (unsigned)(ga & 0xFFFFFFFFu);                     // global_addr lo
    g0[3] = (unsigned)((ga >> 32) & 0x01FFFFFFu) | (2u << 30);// global_addr hi | type=2
    i32x8 g1;
    g1[0] = (int)(1u << 16);                                  // data_size=1 (2 bytes)
    g1[1] = (int)((cols & 0xFFFFu) << 16);                    // tensor_dim0 lo16
    g1[2] = (int)(((cols >> 16) & 0xFFFFu) | ((rows & 0xFFFFu) << 16)); // dim0 hi | dim1 lo
    g1[3] = (int)(((rows >> 16) & 0xFFFFu) | ((cols & 0xFFFFu) << 16)); // dim1 hi | tile_dim0
    g1[4] = (int)(rows & 0xFFFFu);                            // tile_dim1 (tile_dim2=0)
    g1[5] = (int)cols;                                        // tensor_dim0_stride lo32
    g1[6] = 0;
    g1[7] = 0;
    i32x4 z4 = {0, 0, 0, 0};
    i32x8 z8 = {0, 0, 0, 0, 0, 0, 0, 0};
    __builtin_amdgcn_tensor_load_to_lds(g0, g1, z4, z4, z8, 0);
}

// ---------------------------------------------------------------------------
// Kernel 0: f32 -> f16 convert (weights + embedding; L2-resident afterwards)
// ---------------------------------------------------------------------------
__global__ void cvt_f32_f16(const float* __restrict__ in, _Float16* __restrict__ out, int n) {
    int i = blockIdx.x * 256 + threadIdx.x;
    if (i < n) out[i] = (_Float16)in[i];
}

// ---------------------------------------------------------------------------
// Kernel 1: sequential 2-layer LSTM recurrence.
// grid = 16 WGs (16-row batch tile each), block = 256 thr = 8 waves.
// Wave w owns gate N-tiles {w, w+8, w+16, w+24} = complete i/f/g/o for hidden
// units j = w*16 .. w*16+15 -> elementwise phase runs entirely in registers.
// ---------------------------------------------------------------------------
__global__ __launch_bounds__(256) void lstm_seq(
    const int*      __restrict__ src,   // [B,T]
    const int*      __restrict__ lens,  // [B]
    const _Float16* __restrict__ embf,  // [V_IN, E]
    const _Float16* __restrict__ Wx0f,  // [4H, E]   (register-resident frags)
    const float*    __restrict__ bx0,   // [4H]
    const _Float16* __restrict__ Wh0f,  // [4H, H]   (-> LDS via TDM)
    const _Float16* __restrict__ Wx1f,  // [4H, H]   (streamed from L2)
    const float*    __restrict__ bx1,   // [4H]
    const _Float16* __restrict__ Wh1f,  // [4H, H]   (-> LDS via TDM)
    _Float16*       __restrict__ hbuf)  // [B,T,H] f16 h1 history
{
    extern __shared__ _Float16 smem[];
    _Float16* sWh0 = smem + LWH0_;
    _Float16* sWh1 = smem + LWH1_;
    _Float16* sX   = smem + LX_;
    _Float16* sH0  = smem + LH0_;
    _Float16* sH1  = smem + LH1_;

    const int tid  = threadIdx.x;
    const int lane = tid & 31;
    const int w    = tid >> 5;          // wave 0..7
    const int b0   = blockIdx.x * 16;

    const int n    = lane & 15;         // N within tile / D-layout column
    const int mb   = (lane >> 4) << 3;  // D-layout row base (0 or 8)
    const int jcol = w * 16 + n;        // hidden unit owned by this lane

    // ---- one-time: TDM preload of recurrent weights into LDS (wave 0) ----
    if (tid < 32) {
        unsigned ldsbase = (unsigned)(uintptr_t)(void*)smem;
        tdm_load_2d_f16(ldsbase + LWH0_ * 2u, Wh0f, G4_, H_);
        tdm_load_2d_f16(ldsbase + LWH1_ * 2u, Wh1f, G4_, H_);
        __builtin_amdgcn_s_wait_tensorcnt(0);
    }
    // zero h state in LDS
    for (int i = tid; i < 16 * H_; i += 256) {
        sH0[i] = (_Float16)0.f;
        sH1[i] = (_Float16)0.f;
    }
    // gather x_0
    {
        int idx = tid * 4, b = idx >> 6, k = idx & 63;
        int s = src[(size_t)(b0 + b) * T_ + 0];
        *(v4h*)(sX + b * E_ + k) = *(const v4h*)(embf + (size_t)s * E_ + k);
    }

    // per-row sequence lengths for this lane's 8 batch rows
    int lens8[8];
#pragma unroll
    for (int r = 0; r < 8; ++r) lens8[r] = lens[b0 + mb + r];

    // biases for the 4 owned gate tiles (folded into WMMA C-init)
    const float bI0 = bx0[jcol], bF0 = bx0[H_ + jcol], bG0 = bx0[2 * H_ + jcol], bO0 = bx0[3 * H_ + jcol];
    const float bI1 = bx1[jcol], bF1 = bx1[H_ + jcol], bG1 = bx1[2 * H_ + jcol], bO1 = bx1[3 * H_ + jcol];

    float c0[8], c1[8], h0v[8], h1v[8];
#pragma unroll
    for (int r = 0; r < 8; ++r) { c0[r] = c1[r] = h0v[r] = h1v[r] = 0.f; }

    __syncthreads();

    for (int t = 0; t < T_; ++t) {
        // opaque zero: keeps streamed weight-fragment addresses loop-variant so
        // LICM cannot hoist 24 fragments (=192 VGPRs) out of the loop (-> spills).
        int zk;
        asm volatile("v_mov_b32 %0, 0" : "=v"(zk));

        // ---------------- layer 0 gates ----------------
        v8f aI = splat8(bI0), aF = splat8(bF0), aG = splat8(bG0), aO = splat8(bO0);
#pragma unroll
        for (int kt = 0; kt < 2; ++kt) {            // x-part, K=64, Wx0 frags hoisted to regs
            v16h A = loadA(sX, E_, kt * 32, lane);
            aI = wmma(A, loadB_WT(Wx0f, E_, (w     ) * 16, kt * 32, lane), aI);
            aF = wmma(A, loadB_WT(Wx0f, E_, (w +  8) * 16, kt * 32, lane), aF);
            aG = wmma(A, loadB_WT(Wx0f, E_, (w + 16) * 16, kt * 32, lane), aG);
            aO = wmma(A, loadB_WT(Wx0f, E_, (w + 24) * 16, kt * 32, lane), aO);
        }
#pragma unroll
        for (int kt = 0; kt < 4; ++kt) {            // h-part, K=128, Wh0 from LDS
            v16h A = loadA(sH0, H_, kt * 32, lane);
            aI = wmma(A, loadB_WT(sWh0, H_, (w     ) * 16, kt * 32 + zk, lane), aI);
            aF = wmma(A, loadB_WT(sWh0, H_, (w +  8) * 16, kt * 32 + zk, lane), aF);
            aG = wmma(A, loadB_WT(sWh0, H_, (w + 16) * 16, kt * 32 + zk, lane), aG);
            aO = wmma(A, loadB_WT(sWh0, H_, (w + 24) * 16, kt * 32 + zk, lane), aO);
        }
        __syncthreads();                            // all sH0 reads done

        // ---------------- layer 0 nonlinearity (registers only) ----------------
#pragma unroll
        for (int r = 0; r < 8; ++r) {
            float i_ = sigmoidf_(aI[r]), f_ = sigmoidf_(aF[r]);
            float g_ = tanhf(aG[r]),     o_ = sigmoidf_(aO[r]);
            float cn = f_ * c0[r] + i_ * g_;
            float hn = o_ * tanhf(cn);
            bool valid = t < lens8[r];
            c0[r]  = valid ? cn : c0[r];
            h0v[r] = valid ? hn : h0v[r];
            sH0[(mb + r) * H_ + jcol] = (_Float16)h0v[r];
        }
        __syncthreads();                            // sH0 updated

        // ---------------- layer 1 gates ----------------
        aI = splat8(bI1); aF = splat8(bF1); aG = splat8(bG1); aO = splat8(bO1);
#pragma unroll
        for (int kt = 0; kt < 4; ++kt) {            // x-part = h0, Wx1 streamed from L2
            v16h A = loadA(sH0, H_, kt * 32, lane);
            aI = wmma(A, loadB_WT(Wx1f, H_, (w     ) * 16, kt * 32 + zk, lane), aI);
            aF = wmma(A, loadB_WT(Wx1f, H_, (w +  8) * 16, kt * 32 + zk, lane), aF);
            aG = wmma(A, loadB_WT(Wx1f, H_, (w + 16) * 16, kt * 32 + zk, lane), aG);
            aO = wmma(A, loadB_WT(Wx1f, H_, (w + 24) * 16, kt * 32 + zk, lane), aO);
        }
#pragma unroll
        for (int kt = 0; kt < 4; ++kt) {            // h-part, Wh1 from LDS
            v16h A = loadA(sH1, H_, kt * 32, lane);
            aI = wmma(A, loadB_WT(sWh1, H_, (w     ) * 16, kt * 32 + zk, lane), aI);
            aF = wmma(A, loadB_WT(sWh1, H_, (w +  8) * 16, kt * 32 + zk, lane), aF);
            aG = wmma(A, loadB_WT(sWh1, H_, (w + 16) * 16, kt * 32 + zk, lane), aG);
            aO = wmma(A, loadB_WT(sWh1, H_, (w + 24) * 16, kt * 32 + zk, lane), aO);
        }
        __syncthreads();                            // all sH1 reads done

        // ---------------- layer 1 nonlinearity + outputs + prefetch x_{t+1} ----------------
#pragma unroll
        for (int r = 0; r < 8; ++r) {
            float i_ = sigmoidf_(aI[r]), f_ = sigmoidf_(aF[r]);
            float g_ = tanhf(aG[r]),     o_ = sigmoidf_(aO[r]);
            float cn = f_ * c1[r] + i_ * g_;
            float hn = o_ * tanhf(cn);
            bool valid = t < lens8[r];
            c1[r]  = valid ? cn : c1[r];
            h1v[r] = valid ? hn : h1v[r];
            _Float16 hh = (_Float16)h1v[r];
            sH1[(mb + r) * H_ + jcol] = hh;
            hbuf[((size_t)(b0 + mb + r) * T_ + t) * H_ + jcol] = hh;
        }
        if (t + 1 < T_) {                           // gather next x tile (sX free since barrier A)
            int idx = tid * 4, b = idx >> 6, k = idx & 63;
            int s = src[(size_t)(b0 + b) * T_ + (t + 1)];
            *(v4h*)(sX + b * E_ + k) = *(const v4h*)(embf + (size_t)s * E_ + k);
        }
        __syncthreads();
    }
}

// ---------------------------------------------------------------------------
// Kernel 2: logits = h1_history @ Wout^T + bout ; M=131072, N=128, K=128.
// ---------------------------------------------------------------------------
__global__ __launch_bounds__(256) void proj_out(
    const _Float16* __restrict__ hbuf,
    const _Float16* __restrict__ Woutf,
    const float*    __restrict__ bout,
    float*          __restrict__ out)
{
    const int tid   = threadIdx.x;
    const int lane  = tid & 31;
    const int ntile = tid >> 5;          // 0..7
    const int mtile = blockIdx.x;        // 0..8191

    const int n = lane & 15;
    v8f acc = splat8(bout[ntile * 16 + n]);     // bias folded into C-init
    const _Float16* Arow = hbuf + (size_t)mtile * 16 * H_;
#pragma unroll
    for (int kt = 0; kt < 4; ++kt) {
        v16h A = loadA(Arow, H_, kt * 32, lane);
        acc = wmma(A, loadB_WT(Woutf, H_, ntile * 16, kt * 32, lane), acc);
    }
    const int mbr = (lane >> 4) << 3;
#pragma unroll
    for (int r = 0; r < 8; ++r)
        out[((size_t)mtile * 16 + mbr + r) * VOUT_ + ntile * 16 + n] = acc[r];
}

// ---------------------------------------------------------------------------
// Host launcher
// ---------------------------------------------------------------------------
extern "C" void kernel_launch(void* const* d_in, const int* in_sizes, int n_in,
                              void* d_out, int out_size, void* d_ws, size_t ws_size,
                              hipStream_t stream) {
    const int*   src  = (const int*)  d_in[0];
    const int*   lens = (const int*)  d_in[1];
    const float* emb  = (const float*)d_in[2];
    const float* Wx0  = (const float*)d_in[3];
    const float* bx0  = (const float*)d_in[4];
    const float* Wh0  = (const float*)d_in[5];
    const float* Wx1  = (const float*)d_in[6];
    const float* bx1  = (const float*)d_in[7];
    const float* Wh1  = (const float*)d_in[8];
    const float* Wout = (const float*)d_in[9];
    const float* bout = (const float*)d_in[10];
    float* out = (float*)d_out;

    _Float16* w = (_Float16*)d_ws;
    _Float16* embf  = w;               // 4096
    _Float16* Wx0f  = w + 4096;        // 32768
    _Float16* Wh0f  = w + 36864;       // 65536
    _Float16* Wx1f  = w + 102400;      // 65536
    _Float16* Wh1f  = w + 167936;      // 65536
    _Float16* Woutf = w + 233472;      // 16384
    _Float16* hbuf  = w + 249856;      // B*T*H

    auto cvt = [&](const float* in, _Float16* o, int nEl) {
        cvt_f32_f16<<<(nEl + 255) / 256, 256, 0, stream>>>(in, o, nEl);
    };
    cvt(emb,  embf,  64 * E_);
    cvt(Wx0,  Wx0f,  G4_ * E_);
    cvt(Wh0,  Wh0f,  G4_ * H_);
    cvt(Wx1,  Wx1f,  G4_ * H_);
    cvt(Wh1,  Wh1f,  G4_ * H_);
    cvt(Wout, Woutf, VOUT_ * H_);

    (void)hipFuncSetAttribute((const void*)lstm_seq,
                              hipFuncAttributeMaxDynamicSharedMemorySize, LDS_BYTES_);
    lstm_seq<<<B_ / 16, 256, LDS_BYTES_, stream>>>(src, lens, embf, Wx0f, bx0, Wh0f,
                                                   Wx1f, bx1, Wh1f, hbuf);

    proj_out<<<(B_ * T_) / 16, 256, 0, stream>>>(hbuf, Woutf, bout, out);
}
